// Compressor_72395968741435
// MI455X (gfx1250) — compile-verified, matrix operand order checked by reading, not run on hardware
//
#include <hip/hip_runtime.h>
#include <math.h>

// Problem constants (from reference)
#define DDIM   4096      // inner dim
#define ODIM   1024      // coff * head_dim
#define HD     512       // head_dim
#define NTOK   2048      // 2 * nW tokens per batch in output

#define BLOCK  512       // 16 waves
#define M_TILE 64        // rows per workgroup (16 windows of 4)
#define N_TILE 256       // columns per workgroup (per matrix; kv AND gate)
#define K_TILE 32        // bf16 WMMA K depth
#define LDS_STRIDE 36    // 32 + 4 pad (bf16 elems) -> conflict-free b64 reads
#define NKSTEP (DDIM / K_TILE)   // 128

typedef __attribute__((ext_vector_type(16))) __bf16 v16bf;
typedef __attribute__((ext_vector_type(8)))  float  v8f;

union Frag { unsigned int u[8]; v16bf v; };

#if defined(__has_builtin)
#if __has_builtin(__builtin_amdgcn_cvt_pk_bf16_f32)
#define HAVE_CVT_PK_BF16 1
#endif
#endif

// Pack two fp32 -> packed 2x bf16 in one instruction.
__device__ __forceinline__ unsigned int pack2bf(float a, float b) {
#ifdef HAVE_CVT_PK_BF16
  auto r = __builtin_amdgcn_cvt_pk_bf16_f32(a, b);   // v_cvt_pk_bf16_f32
  return __builtin_bit_cast(unsigned int, r);
#else
  // v_perm_b32: result bytes = {a.b2, a.b3, b.b2, b.b3} = {bf16(a), bf16(b)}
  // (truncating bf16 conversion; error is same order as bf16 quantization)
  unsigned int ua = __builtin_bit_cast(unsigned int, a);
  unsigned int ub = __builtin_bit_cast(unsigned int, b);
  return __builtin_amdgcn_perm(ub, ua, 0x07060302u);
#endif
}

// pack 4 floats -> 4 bf16, one 8B LDS store (address 8B aligned by construction)
__device__ __forceinline__ void st8(unsigned short* p, float4 v) {
  uint2 t;
  t.x = pack2bf(v.x, v.y);
  t.y = pack2bf(v.z, v.w);
  *(uint2*)p = t;
}

// Load a 16x32 bf16 fragment (A layout; B uses the mirrored layout with N on lanes)
__device__ __forceinline__ Frag load_frag(const unsigned short* base, int row, int half) {
  Frag f;
  const unsigned short* p = base + row * LDS_STRIDE + half * 8;
  uint2 q0 = *(const uint2*)(p);           // K = h*8 + 0..3
  uint2 q1 = *(const uint2*)(p + 4);       // K = h*8 + 4..7
  uint2 q2 = *(const uint2*)(p + 16);      // K = 16 + h*8 + 0..3
  uint2 q3 = *(const uint2*)(p + 20);      // K = 16 + h*8 + 4..7
  f.u[0] = q0.x; f.u[1] = q0.y;
  f.u[2] = q1.x; f.u[3] = q1.y;
  f.u[4] = q2.x; f.u[5] = q2.y;
  f.u[6] = q3.x; f.u[7] = q3.y;
  return f;
}

__device__ __forceinline__ v8f wmma_bf16(const Frag& a, const Frag& b, v8f c) {
  return __builtin_amdgcn_wmma_f32_16x16x32_bf16(
      /*neg_a=*/false, a.v, /*neg_b=*/false, b.v,
      /*c_mod=*/(short)0, c, /*reuse_a=*/false, /*reuse_b=*/false);
}

__device__ __forceinline__ void load_stage(const float* __restrict__ x,
                                           const float* __restrict__ wkv,
                                           const float* __restrict__ wg,
                                           int m0, int n0, int k0, int tid,
                                           float4& a0, float4* rk, float4* rg) {
  const int ar = tid >> 3, aq = tid & 7;
  const float4* xv  = (const float4*)x;
  const float4* kv4 = (const float4*)wkv;
  const float4* g4  = (const float4*)wg;
  a0 = xv[(((m0 + ar) * DDIM + k0) >> 2) + aq];
#pragma unroll
  for (int j = 0; j < 4; ++j) {
    int idx = tid + j * BLOCK;
    int n = idx >> 3, q = idx & 7;
    rk[j] = kv4[(((n0 + n) * DDIM + k0) >> 2) + q];
    rg[j] =  g4[(((n0 + n) * DDIM + k0) >> 2) + q];
  }
}

__device__ __forceinline__ void store_stage(unsigned short* sA, unsigned short* sK,
                                            unsigned short* sG, int tid,
                                            float4 a0, const float4* rk, const float4* rg) {
  const int ar = tid >> 3, aq = tid & 7;
  st8(&sA[ar * LDS_STRIDE + aq * 4], a0);
#pragma unroll
  for (int j = 0; j < 4; ++j) {
    int idx = tid + j * BLOCK;
    int n = idx >> 3, q = idx & 7;
    st8(&sK[n * LDS_STRIDE + q * 4], rk[j]);
    st8(&sG[n * LDS_STRIDE + q * 4], rg[j]);
  }
}

__global__ __launch_bounds__(BLOCK)
void nsa_compress_gemm(const float* __restrict__ x,
                       const float* __restrict__ wkv,
                       const float* __restrict__ wg,
                       const float* __restrict__ ape,
                       float* __restrict__ out) {
  __shared__ unsigned short ldsA[M_TILE * LDS_STRIDE];   //  4.5 KB
  __shared__ unsigned short ldsK[N_TILE * LDS_STRIDE];   // 18.4 KB
  __shared__ unsigned short ldsG[N_TILE * LDS_STRIDE];   // 18.4 KB

  const int tid  = threadIdx.x;
  const int wave = tid >> 5;        // 0..15 -> 16-column slice of N_TILE
  const int lane = tid & 31;
  const int half = lane >> 4;
  const int l16  = lane & 15;

  const int m0 = blockIdx.x * M_TILE;   // global row (b*S + s) tile base
  const int n0 = blockIdx.y * N_TILE;   // column tile base within ODIM

  const int ocol = n0 + wave * 16 + l16;          // this lane's output column
  const float pe0 = ape[0 * ODIM + ocol];
  const float pe1 = ape[1 * ODIM + ocol];
  const float pe2 = ape[2 * ODIM + ocol];
  const float pe3 = ape[3 * ODIM + ocol];

  v8f acc_kv[4], acc_g[4];
#pragma unroll
  for (int i = 0; i < 4; ++i)
#pragma unroll
    for (int j = 0; j < 8; ++j) { acc_kv[i][j] = 0.f; acc_g[i][j] = 0.f; }

  float4 a0, rk[4], rg[4];
  load_stage(x, wkv, wg, m0, n0, 0, tid, a0, rk, rg);

  for (int kt = 0; kt < NKSTEP; ++kt) {
    // write the staged K-step into LDS (bf16)
    store_stage(ldsA, ldsK, ldsG, tid, a0, rk, rg);
    __syncthreads();

    // start global loads for the next K-step (overlaps WMMA below)
    if (kt + 1 < NKSTEP)
      load_stage(x, wkv, wg, m0, n0, (kt + 1) * K_TILE, tid, a0, rk, rg);

    // prefetch two K-steps ahead into cache (global_prefetch_b8)
    if (kt + 2 < NKSTEP) {
      const int k2 = (kt + 2) * K_TILE;
      __builtin_prefetch(&x[(m0 + (tid >> 3)) * DDIM + k2 + ((tid & 7) << 2)], 0, 1);
      __builtin_prefetch(&wkv[(n0 + (tid >> 1)) * DDIM + k2 + ((tid & 1) << 4)], 0, 1);
      __builtin_prefetch(&wg [(n0 + (tid >> 1)) * DDIM + k2 + ((tid & 1) << 4)], 0, 1);
    }

    // this wave's B fragments (shared across the 4 M-subtiles)
    Frag bk = load_frag(ldsK, wave * 16 + l16, half);
    Frag bg = load_frag(ldsG, wave * 16 + l16, half);
#pragma unroll
    for (int mt = 0; mt < 4; ++mt) {
      Frag a = load_frag(ldsA, mt * 16 + l16, half);
      acc_kv[mt] = wmma_bf16(a, bk, acc_kv[mt]);
      acc_g[mt]  = wmma_bf16(a, bg, acc_g[mt]);
    }
    __syncthreads();   // all waves done reading LDS before next store
  }

  // Epilogue: sigmoid gate, +ape, window-of-4 mean pool, write pre-norm values.
  const int c = ocol >> 9;           // overlap stream (0/1)
  const int h = ocol & (HD - 1);     // head-dim index
#pragma unroll
  for (int mt = 0; mt < 4; ++mt) {
    float g[8];
#pragma unroll
    for (int r = 0; r < 8; ++r) {
      float kvv = acc_kv[mt][r];
      float gv  = acc_g[mt][r];
      float sig = 1.0f / (1.0f + __expf(-gv));
      float pe  = ((r & 3) == 0) ? pe0 : ((r & 3) == 1) ? pe1 : ((r & 3) == 2) ? pe2 : pe3;
      g[r] = kvv * sig + pe;
    }
    float p0 = (g[0] + g[1] + g[2] + g[3]) * 0.25f;   // window mt*4 + half*2
    float p1 = (g[4] + g[5] + g[6] + g[7]) * 0.25f;   // window mt*4 + half*2 + 1

    int wing = (m0 >> 2) + mt * 4 + half * 2;   // global window id (0..4095)
    int b    = wing >> 10;                      // 1024 windows per batch
    int winb = wing & 1023;
    int base = ((b * NTOK) + winb * 2 + c) * HD + h;
    out[base]          = p0;
    out[base + 2 * HD] = p1;                    // next window -> token index +2
  }
}

__global__ __launch_bounds__(256)
void rmsnorm_inplace(float* __restrict__ out, const float* __restrict__ nw) {
  const int wave = threadIdx.x >> 5;
  const int lane = threadIdx.x & 31;
  const int row  = blockIdx.x * 8 + wave;   // 8192 rows of 512
  float* p = out + row * HD;

  float v[16];
  float s = 0.f;
#pragma unroll
  for (int j = 0; j < 16; ++j) { v[j] = p[j * 32 + lane]; s += v[j] * v[j]; }
#pragma unroll
  for (int off = 16; off > 0; off >>= 1) s += __shfl_xor(s, off, 32);
  const float scale = rsqrtf(s * (1.0f / HD) + 1e-6f);
#pragma unroll
  for (int j = 0; j < 16; ++j) p[j * 32 + lane] = v[j] * scale * nw[j * 32 + lane];
}

extern "C" void kernel_launch(void* const* d_in, const int* in_sizes, int n_in,
                              void* d_out, int out_size, void* d_ws, size_t ws_size,
                              hipStream_t stream) {
  (void)in_sizes; (void)n_in; (void)out_size; (void)d_ws; (void)ws_size;
  const float* x   = (const float*)d_in[0];
  const float* wkv = (const float*)d_in[1];
  const float* wg  = (const float*)d_in[2];
  const float* ape = (const float*)d_in[3];
  const float* nw  = (const float*)d_in[4];
  float* out = (float*)d_out;

  dim3 grid1(16384 / M_TILE, ODIM / N_TILE);   // 256 x 4
  nsa_compress_gemm<<<grid1, dim3(BLOCK), 0, stream>>>(x, wkv, wg, ape, out);
  rmsnorm_inplace<<<dim3(8192 / 8), dim3(256), 0, stream>>>(out, nw);
}
